// escargot_33397665694544
// MI455X (gfx1250) — matrix operand order repaired and checked by reading, not compiled, not run on hardware
//
#include <hip/hip_runtime.h>

// ---------------------------------------------------------------------------
// Types for CDNA5 WMMA (wave32): v_wmma_f32_16x16x32_bf16
// ---------------------------------------------------------------------------
typedef __attribute__((ext_vector_type(16))) __bf16 v16bf;
typedef __attribute__((ext_vector_type(8)))  float  v8f;

union AFrag { v16bf v; unsigned int u[8]; };

__device__ __forceinline__ v8f vzero8() {
  v8f z;
#pragma unroll
  for (int i = 0; i < 8; ++i) z[i] = 0.0f;
  return z;
}

__device__ __forceinline__ v8f wmma_bf16(v16bf a, v16bf b, v8f c) {
  // 8 args: (neg_a, A, neg_b, B, c_mod, C, reuse_a, reuse_b)
  return __builtin_amdgcn_wmma_f32_16x16x32_bf16(false, a, false, b, (short)0, c,
                                                 false, false);
}

// A-fragment (16x32 bf16, contiguous K order matching B):
//  lanes 0-15 : m = lane,    K = {0..15}
//  lanes 16-31: m = lane-16, K = {16..31}
// Pairs (K, K+1) are contiguous -> 8x 32-bit LDS loads from a row-major tile.
__device__ __forceinline__ v16bf load_afrag(const __bf16* rowptr, int kb, int hlf) {
  AFrag a;
#pragma unroll
  for (int j2 = 0; j2 < 8; ++j2) {
    const int k0 = 2 * j2 + hlf * 16;
    a.u[j2] = *(const unsigned int*)(rowptr + kb + k0);
  }
  return a.v;
}

__device__ __forceinline__ float silu_f(float x) {
  return x * (1.0f / (1.0f + __expf(-x)));
}

// Uniform cubic B-spline basis: knots t_j = 0.4*j - 2.2, j = 0..11 -> 8 funcs.
// Closed-form cardinal cubic (grid is uniform, so Cox-de Boor collapses).
__device__ __forceinline__ void bspline8(float x, float b[8]) {
  const float s  = (x + 2.2f) * 2.5f;
  const float fi = floorf(s);
  const int   i  = (int)fi;
  const float u  = s - fi;
  const float u2 = u * u, u3 = u2 * u;
  const float om = 1.0f - u;
  const float v0 = u3 * (1.0f / 6.0f);                                 // j = i
  const float v1 = (-3.0f * u3 + 3.0f * u2 + 3.0f * u + 1.0f) * (1.0f / 6.0f);
  const float v2 = (3.0f * u3 - 6.0f * u2 + 4.0f) * (1.0f / 6.0f);
  const float v3 = om * om * om * (1.0f / 6.0f);                       // j = i-3
#pragma unroll
  for (int c = 0; c < 8; ++c) {
    const int d = i - c;
    float r = 0.0f;
    r = (d == 0) ? v0 : r;
    r = (d == 1) ? v1 : r;
    r = (d == 2) ? v2 : r;
    r = (d == 3) ? v3 : r;
    b[c] = r;
  }
}

// ---------------------------------------------------------------------------
// Problem geometry
// ---------------------------------------------------------------------------
#define NBATCH 2048
#define INF1   529      // layer1 in features
#define K1TOT  4896     // 544 padded features * 9 (silu + 8 basis)
#define K1CH   288      // 32 features * 9 per chunk (9 WMMA k-steps)
#define NCH1   17       // ceil(529/32)
#define K2TOT  384      // 40*9 = 360 padded to 384 (12 WMMA k-steps)
#define ABUF_S 388      // LDS row stride (>= max(288,384), even)
#define K3TOT  960      // head GEMM K (30 k-steps)

// ---------------------------------------------------------------------------
// Kernel 0: pack combined bf16 weights  [base | spline] per KAN layer
// ---------------------------------------------------------------------------
__global__ void pack_weights(const float* __restrict__ base_w1,
                             const float* __restrict__ spline_w1,
                             const float* __restrict__ base_w2,
                             const float* __restrict__ spline_w2,
                             const float* __restrict__ lin1_w,
                             const float* __restrict__ lin2_w,
                             __bf16* __restrict__ Wc1,   // (48, 4896)
                             __bf16* __restrict__ Wc2,   // (64, 384)
                             __bf16* __restrict__ W3,    // (512, 960)
                             __bf16* __restrict__ W4) {  // (16, 512)
  const int idx    = blockIdx.x * blockDim.x + threadIdx.x;
  const int stride = gridDim.x * blockDim.x;
  for (int p = idx; p < 48 * K1TOT; p += stride) {
    const int n = p / K1TOT, k = p % K1TOT;
    const int f = k / 9, c = k % 9;
    float v = 0.0f;
    if (n < 40 && f < INF1)
      v = (c == 0) ? base_w1[n * INF1 + f] : spline_w1[(n * INF1 + f) * 8 + (c - 1)];
    Wc1[p] = (__bf16)v;
  }
  for (int p = idx; p < 64 * K2TOT; p += stride) {
    const int n = p / K2TOT, k = p % K2TOT;
    const int f = k / 9, c = k % 9;
    float v = 0.0f;
    if (n < 60 && k < 360)
      v = (c == 0) ? base_w2[n * 40 + f] : spline_w2[(n * 40 + f) * 8 + (c - 1)];
    Wc2[p] = (__bf16)v;
  }
  for (int p = idx; p < 512 * K3TOT; p += stride) W3[p] = (__bf16)lin1_w[p];
  for (int p = idx; p < 16 * 512;   p += stride) W4[p] = (__bf16)lin2_w[p];
}

// ---------------------------------------------------------------------------
// Kernel 1: fused KAN layer1 (529->40) + layer2 (40->60).
// Row order r = n*16 + w : one M-tile (16 rows) = all 16 w-slices of one n,
// so each 32-feature chunk of x is a contiguous 2KB block. Output goes
// directly into the (2048, 960) head layout: h960[n][c*16 + w].
// 4 waves / block, one M-tile per wave, per-wave LDS staging (sync'd).
// ---------------------------------------------------------------------------
__global__ __launch_bounds__(128) void kan12_kernel(
    const float* __restrict__ x,     // (2048, 529, 16)
    const __bf16* __restrict__ Wc1,  // (48, 4896)
    const __bf16* __restrict__ Wc2,  // (64, 384)
    float* __restrict__ h960) {      // (2048, 960)
  __shared__ __bf16 ABUF[4][16][ABUF_S];  // augmented A tile (layer1 then layer2)
  __shared__ float  H1[4][16][40];        // layer1 result staging

  const int tid  = threadIdx.x;
  const int lane = tid & 31;
  const int wv   = tid >> 5;
  const int hlf  = lane >> 4;
  const int lm   = lane & 15;
  const int nb   = blockIdx.x * 4 + wv;          // batch index n (0..2047)
  const float* xrow = x + (size_t)nb * (INF1 * 16);

  v8f c1[3];
#pragma unroll
  for (int t = 0; t < 3; ++t) c1[t] = vzero8();

  for (int ch = 0; ch < NCH1; ++ch) {
    const int fbase = ch * 32;
    if (ch + 1 < NCH1)
      __builtin_prefetch(xrow + (fbase + 32) * 16, 0, 1);
    // Build augmented A rows: per feature f -> [silu(x), basis0..7] as bf16.
#pragma unroll 4
    for (int q = 0; q < 16; ++q) {
      const int vi = lane + 32 * q;          // 0..511, coalesced over lanes
      const int lf = vi >> 4;                // local feature 0..31
      const int w  = vi & 15;                // row (= w slice)
      const int f  = fbase + lf;
      float bas[8];
      float sv = 0.0f;
      if (f < INF1) {
        const float xv = xrow[f * 16 + w];
        sv = silu_f(xv);
        bspline8(xv, bas);
      } else {
#pragma unroll
        for (int c = 0; c < 8; ++c) bas[c] = 0.0f;
      }
      __bf16* dst = &ABUF[wv][w][lf * 9];
      dst[0] = (__bf16)sv;
#pragma unroll
      for (int c = 0; c < 8; ++c) dst[1 + c] = (__bf16)bas[c];
    }
    __syncthreads();
#pragma unroll
    for (int ks = 0; ks < 9; ++ks) {
      const int kb = ks * 32;
      const v16bf a = load_afrag(&ABUF[wv][lm][0], kb, hlf);
      const int kg = ch * K1CH + kb + hlf * 16;   // 32B-aligned global K offset
#pragma unroll
      for (int t = 0; t < 3; ++t) {
        const int n = t * 16 + lm;                 // B column
        const v16bf b = *(const v16bf*)&Wc1[(size_t)n * K1TOT + kg];
        c1[t] = wmma_bf16(a, b, c1[t]);
      }
    }
    __syncthreads();
  }

  // Stage layer1 result (16 x 40) in LDS.
#pragma unroll
  for (int t = 0; t < 3; ++t) {
    const int n = t * 16 + lm;
    if (n < 40) {
#pragma unroll
      for (int i = 0; i < 8; ++i) H1[wv][i + hlf * 8][n] = c1[t][i];
    }
  }
  __syncthreads();

  // Build layer2 augmented A (16 x 384) reusing ABUF.
#pragma unroll 4
  for (int q = 0; q < 20; ++q) {
    const int vi = lane + 32 * q;    // 0..639
    const int m  = vi / 40;
    const int f  = vi % 40;
    const float hv = H1[wv][m][f];
    float bas[8];
    const float sv = silu_f(hv);
    bspline8(hv, bas);
    __bf16* dst = &ABUF[wv][m][f * 9];
    dst[0] = (__bf16)sv;
#pragma unroll
    for (int c = 0; c < 8; ++c) dst[1 + c] = (__bf16)bas[c];
  }
#pragma unroll
  for (int q = 0; q < 12; ++q) {     // zero pad K 360..383
    const int vi = lane + 32 * q;    // 0..383
    const int m  = vi / 24;
    const int k  = 360 + (vi % 24);
    ABUF[wv][m][k] = (__bf16)0.0f;
  }
  __syncthreads();

  v8f c2[4];
#pragma unroll
  for (int t = 0; t < 4; ++t) c2[t] = vzero8();
#pragma unroll
  for (int ks = 0; ks < 12; ++ks) {
    const int kb = ks * 32;
    const v16bf a = load_afrag(&ABUF[wv][lm][0], kb, hlf);
#pragma unroll
    for (int t = 0; t < 4; ++t) {
      const int n = t * 16 + lm;
      const v16bf b = *(const v16bf*)&Wc2[(size_t)n * K2TOT + kb + hlf * 16];
      c2[t] = wmma_bf16(a, b, c2[t]);
    }
  }

  // Scatter into head layout: h960[n][c*16 + w], rows m == w.
#pragma unroll
  for (int t = 0; t < 4; ++t) {
    const int cc = t * 16 + lm;
    if (cc < 60) {
#pragma unroll
      for (int i = 0; i < 8; ++i) {
        const int m = i + hlf * 8;
        h960[(size_t)nb * 960 + cc * 16 + m] = c2[t][i];
      }
    }
  }
}

// ---------------------------------------------------------------------------
// Kernel 2: head. Per block: 16 batch rows.
//   h512 = l2norm(h960 @ W3^T + b1); g = act_base*silu(h512) + basis@act_coeff
//   out  = l2norm(g @ W4^T + b2)
// 4 waves split the 512 output columns of the big GEMM (8 N-tiles each).
// ---------------------------------------------------------------------------
__global__ __launch_bounds__(128) void head_kernel(
    const float* __restrict__ h960,       // (2048, 960)
    const __bf16* __restrict__ W3,        // (512, 960)
    const float* __restrict__ lin1_b,     // (512)
    const float* __restrict__ act_base,   // (512)
    const float* __restrict__ act_coeff,  // (512, 8)
    const __bf16* __restrict__ W4,        // (16, 512)
    const float* __restrict__ lin2_b,     // (16)
    float* __restrict__ out) {            // (2048, 16)
  union HeadU {
    __bf16 a[16][964];   // bf16 A tile for big GEMM
    float  h[16][520];   // f32 h512 (reuses same LDS after GEMM)
  };
  __shared__ HeadU  U;
  __shared__ __bf16 Gbuf[16][516];
  __shared__ float  partial[128];
  __shared__ float  rowinv[16];
  __shared__ float  O[16][17];

  const int tid  = threadIdx.x;
  const int lane = tid & 31;
  const int wv   = tid >> 5;
  const int hlf  = lane >> 4;
  const int lm   = lane & 15;
  const int mb   = blockIdx.x;   // 0..127 -> rows mb*16 .. mb*16+15

  // 1) Load + convert A tile (16 x 960) to bf16.
  for (int p = tid; p < 16 * K3TOT; p += 128) {
    const int m = p / K3TOT, k = p % K3TOT;
    U.a[m][k] = (__bf16)h960[(size_t)(mb * 16 + m) * K3TOT + k];
  }
  __syncthreads();

  // 2) GEMM: wave wv covers columns [wv*128, wv*128+128).
  v8f acc[8];
#pragma unroll
  for (int t = 0; t < 8; ++t) acc[t] = vzero8();
  for (int ks = 0; ks < 30; ++ks) {
    const int kb = ks * 32;
    const v16bf a = load_afrag(&U.a[lm][0], kb, hlf);
#pragma unroll
    for (int t = 0; t < 8; ++t) {
      const int ccol = wv * 128 + t * 16 + lm;
      const v16bf b = *(const v16bf*)&W3[(size_t)ccol * K3TOT + kb + hlf * 16];
      acc[t] = wmma_bf16(a, b, acc[t]);
    }
  }
  __syncthreads();  // done reading U.a (all waves) before overwriting as U.h

  // 3) h512 + bias -> LDS f32.
#pragma unroll
  for (int t = 0; t < 8; ++t) {
    const int ccol = wv * 128 + t * 16 + lm;
#pragma unroll
    for (int i = 0; i < 8; ++i)
      U.h[i + hlf * 8][ccol] = acc[t][i] + lin1_b[ccol];
  }
  __syncthreads();

  // 4) Row L2 norms (8 partial sums of 64 per row).
  {
    const int m = tid & 15, g = tid >> 4;
    float ss = 0.0f;
    for (int j = 0; j < 64; ++j) {
      const float v = U.h[m][g * 64 + j];
      ss += v * v;
    }
    partial[tid] = ss;
  }
  __syncthreads();
  if (tid < 16) {
    float ss = 0.0f;
#pragma unroll
    for (int g = 0; g < 8; ++g) ss += partial[g * 16 + tid];
    rowinv[tid] = 1.0f / fmaxf(sqrtf(ss), 1e-12f);
  }
  __syncthreads();

  // 5) Per-dim spline activation -> bf16 G tile.
  {
    const int m = tid & 15, g = tid >> 4;
    const float inv = rowinv[m];
    for (int j = 0; j < 64; ++j) {
      const int d = g * 64 + j;
      const float hn = U.h[m][d] * inv;
      float bas[8];
      bspline8(hn, bas);
      float gv = act_base[d] * silu_f(hn);
#pragma unroll
      for (int c = 0; c < 8; ++c) gv += bas[c] * act_coeff[d * 8 + c];
      Gbuf[m][d] = (__bf16)gv;
    }
  }
  __syncthreads();

  // 6) lin2 (512 -> 16) via WMMA on wave 0.
  if (wv == 0) {
    v8f o = vzero8();
#pragma unroll
    for (int ks = 0; ks < 16; ++ks) {
      const int kb = ks * 32;
      const v16bf a = load_afrag(&Gbuf[lm][0], kb, hlf);
      const v16bf b = *(const v16bf*)&W4[(size_t)lm * 512 + kb + hlf * 16];
      o = wmma_bf16(a, b, o);
    }
#pragma unroll
    for (int i = 0; i < 8; ++i)
      O[i + hlf * 8][lm] = o[i] + lin2_b[lm];
  }
  __syncthreads();

  // 7) Final L2 norm + store.
  if (tid < 16) {
    float ss = 0.0f;
#pragma unroll
    for (int j = 0; j < 16; ++j) {
      const float v = O[tid][j];
      ss += v * v;
    }
    const float inv = 1.0f / fmaxf(sqrtf(ss), 1e-12f);
#pragma unroll
    for (int j = 0; j < 16; ++j)
      out[(size_t)(mb * 16 + tid) * 16 + j] = O[tid][j] * inv;
  }
}

// ---------------------------------------------------------------------------
// Launch
// ---------------------------------------------------------------------------
extern "C" void kernel_launch(void* const* d_in, const int* in_sizes, int n_in,
                              void* d_out, int out_size, void* d_ws, size_t ws_size,
                              hipStream_t stream) {
  (void)in_sizes; (void)n_in; (void)out_size; (void)ws_size;
  const float* x         = (const float*)d_in[0];
  // d_in[1] xvaw: unused by the reference
  // d_in[2] grid1, d_in[5] grid2, d_in[10] act_grid: uniform grids, hardcoded
  const float* base_w1   = (const float*)d_in[3];
  const float* spline_w1 = (const float*)d_in[4];
  const float* base_w2   = (const float*)d_in[6];
  const float* spline_w2 = (const float*)d_in[7];
  const float* lin1_w    = (const float*)d_in[8];
  const float* lin1_b    = (const float*)d_in[9];
  const float* act_base  = (const float*)d_in[11];
  const float* act_coeff = (const float*)d_in[12];
  const float* lin2_w    = (const float*)d_in[13];
  const float* lin2_b    = (const float*)d_in[14];
  float* out = (float*)d_out;

  // Workspace carve-out (256B aligned), total ~9.4 MB.
  char* ws = (char*)d_ws;
  size_t o = 0;
  auto take = [&](size_t bytes) {
    size_t cur = o;
    o += (bytes + 255) & ~(size_t)255;
    return cur;
  };
  __bf16* Wc1  = (__bf16*)(ws + take((size_t)48 * K1TOT * 2));
  __bf16* Wc2  = (__bf16*)(ws + take((size_t)64 * K2TOT * 2));
  __bf16* W3   = (__bf16*)(ws + take((size_t)512 * K3TOT * 2));
  __bf16* W4   = (__bf16*)(ws + take((size_t)16 * 512 * 2));
  float*  h960 = (float*)(ws + take((size_t)NBATCH * 960 * 4));

  pack_weights<<<512, 256, 0, stream>>>(base_w1, spline_w1, base_w2, spline_w2,
                                        lin1_w, lin2_w, Wc1, Wc2, W3, W4);
  kan12_kernel<<<NBATCH / 4, 128, 0, stream>>>(x, Wc1, Wc2, h960);
  head_kernel<<<NBATCH / 16, 128, 0, stream>>>(h960, W3, lin1_b, act_base,
                                               act_coeff, W4, lin2_b, out);
}
